// StageNet_2078764171306
// MI455X (gfx1250) — matrix operand order repaired and checked by reading, not compiled, not run on hardware
//
#include <hip/hip_runtime.h>
#include <hip/hip_bf16.h>

typedef __attribute__((ext_vector_type(16))) _Float16 v16h;
typedef __attribute__((ext_vector_type(8)))  float    v8f;

#define B_   128
#define T_   50
#define V_   4000
#define D_   128
#define H_   384
#define L_   3
#define C_   128
#define KD_  10        // CONV depth
#define NO_  1542      // 4H + 2L
#define NOP_ 1600      // NO_ padded to 100 tiles of 16
#define NT_O 100
#define NT_H 24        // 384/16
#define NT_S 4         // 64/16
#define OUT_ 128
#define HS_  64        // H/6

// ---------------- WMMA tile helpers ----------------

// A 16x32 f16 tile, row-major source: one contiguous 16-half run per lane (K = 16g .. 16g+15).
__device__ inline v16h load_a_tile(const _Float16* __restrict__ Arow0, int lda, int k0) {
  int lane = threadIdx.x & 31;
  int g = lane >> 4, m = lane & 15;
  const _Float16* p = Arow0 + (size_t)m * lda + k0 + 16 * g;
  v16h a;
#pragma unroll
  for (int e = 0; e < 16; ++e) a[e] = p[e];         // K = 16g .. 16g+15
  return a;
}

// Pre-swizzled B: [kt*NT+nt][lane][16 halfs] -> one aligned v16h load per tile.
__device__ inline v16h load_b_packed(const _Float16* __restrict__ Bpk, int kt, int NT, int nt) {
  size_t idx = (((size_t)(kt * NT + nt) * 32 + (threadIdx.x & 31)) << 4);
  return *(const v16h*)(Bpk + idx);
}

__device__ inline float sigmoidf_(float x) { return 1.0f / (1.0f + expf(-x)); }

#define WMMA_(A, Bv, Cv) __builtin_amdgcn_wmma_f32_16x16x32_f16(false, (A), false, (Bv), (short)0, (Cv), false, false)

// ---------------- prep kernels ----------------

__global__ void k_zero_f32(float* __restrict__ p, int n) {
  for (int i = blockIdx.x * blockDim.x + threadIdx.x; i < n; i += gridDim.x * blockDim.x)
    p[i] = 0.0f;
}
__global__ void k_zero_f16(_Float16* __restrict__ p, int n) {
  for (int i = blockIdx.x * blockDim.x + threadIdx.x; i < n; i += gridDim.x * blockDim.x)
    p[i] = (_Float16)0.0f;
}

// Pack row-major f32 B[k][col] (ld = Ncols) into WMMA tile-swizzled f16, N padded to NT*16.
__global__ void k_pack_b(_Float16* __restrict__ dst, const float* __restrict__ src,
                         int Kdim, int Ncols, int NT) {
  int total = (Kdim >> 5) * NT * 512;
  for (int i = blockIdx.x * blockDim.x + threadIdx.x; i < total; i += gridDim.x * blockDim.x) {
    int e = i & 15, lane = (i >> 4) & 31, tile = i >> 9;
    int nt = tile % NT, kt = tile / NT;
    int g = lane >> 4, n = lane & 15;
    int k = kt * 32 + e + 16 * g;
    int col = nt * 16 + n;
    dst[i] = (col < Ncols) ? (_Float16)src[(size_t)k * Ncols + col] : (_Float16)0.0f;
  }
}
// Same, but source is transposed: B[k][col] = src[col * Kdim + k]  (used for Wconv)
__global__ void k_pack_bT(_Float16* __restrict__ dst, const float* __restrict__ src,
                          int Kdim, int Ncols, int NT) {
  int total = (Kdim >> 5) * NT * 512;
  for (int i = blockIdx.x * blockDim.x + threadIdx.x; i < total; i += gridDim.x * blockDim.x) {
    int e = i & 15, lane = (i >> 4) & 31, tile = i >> 9;
    int nt = tile % NT, kt = tile / NT;
    int g = lane >> 4, n = lane & 15;
    int k = kt * 32 + e + 16 * g;
    int col = nt * 16 + n;
    dst[i] = (col < Ncols) ? (_Float16)src[(size_t)col * Kdim + k] : (_Float16)0.0f;
  }
}

__global__ void k_build_bo(float* __restrict__ bo, const float* __restrict__ bk,
                           const float* __restrict__ br, const float* __restrict__ wk_last,
                           const float* __restrict__ wr_last) {
  int i = blockIdx.x * blockDim.x + threadIdx.x;
  if (i < NOP_) bo[i] = (i < NO_) ? (bk[i] + br[i] + wk_last[i] + wr_last[i]) : 0.0f;
}

// ---------------- embedding ----------------

__global__ __launch_bounds__(128) void k_embed(_Float16* __restrict__ x16,
                                               int* __restrict__ mask,
                                               const int* __restrict__ multihot,
                                               const float* __restrict__ emb) {
  int bt = blockIdx.x;            // b*T + t
  int d = threadIdx.x;
  __shared__ int s_mh[128];
  __shared__ int s_any;
  if (d == 0) s_any = 0;
  const int* mrow = multihot + (size_t)bt * V_;
  float acc = 0.0f;
  for (int v0 = 0; v0 < V_; v0 += 128) {
    int idx = v0 + d;
    int mv = (idx < V_) ? mrow[idx] : 0;
    __syncthreads();
    s_mh[d] = mv;
    __syncthreads();
    int lim = (V_ - v0 < 128) ? (V_ - v0) : 128;
    for (int v = 0; v < lim; ++v) {
      if (s_mh[v]) acc += emb[(size_t)(v0 + v) * D_ + d];
    }
  }
  x16[(size_t)bt * D_ + d] = (_Float16)acc;
  if (acc != 0.0f) atomicOr(&s_any, 1);
  __syncthreads();
  if (d == 0) mask[bt] = s_any;
}

__global__ void k_last(int* __restrict__ last, const int* __restrict__ mask) {
  int b = threadIdx.x;   // 128 threads
  int s = 0;
  for (int t = 0; t < T_; ++t) s += mask[b * T_ + t];
  last[b] = s - 1;
}

// ---------------- per-step GEMMs: one wave per 16x64 output strip ----------------

// o[128][NOP_] = x_t @ Wk + h @ Wr + bo
__global__ __launch_bounds__(32) void k_gemm_o(float* __restrict__ o,
    const _Float16* __restrict__ xrow, int ldax,
    const _Float16* __restrict__ WkPk, const _Float16* __restrict__ h16,
    const _Float16* __restrict__ WrPk, const float* __restrict__ bo) {
  int nt0 = blockIdx.x * 4;      // gridDim.x = 25
  int m0 = blockIdx.y * 16;
  v8f acc[4] = {};
  for (int kt = 0; kt < (D_ >> 5); ++kt) {
    v16h a = load_a_tile(xrow + (size_t)m0 * ldax, ldax, kt * 32);
#pragma unroll
    for (int i = 0; i < 4; ++i)
      acc[i] = WMMA_(a, load_b_packed(WkPk, kt, NT_O, nt0 + i), acc[i]);
  }
  for (int kt = 0; kt < (H_ >> 5); ++kt) {
    v16h a = load_a_tile(h16 + (size_t)m0 * H_, H_, kt * 32);
#pragma unroll
    for (int i = 0; i < 4; ++i)
      acc[i] = WMMA_(a, load_b_packed(WrPk, kt, NT_O, nt0 + i), acc[i]);
  }
  int lane = threadIdx.x, g = lane >> 4, n = lane & 15;
#pragma unroll
  for (int i = 0; i < 4; ++i) {
    int col = (nt0 + i) * 16 + n;
    float bias = bo[col];
#pragma unroll
    for (int r = 0; r < 8; ++r)
      o[(size_t)(m0 + r + 8 * g) * NOP_ + col] = acc[i][r] + bias;
  }
}

// gates / LSTM-like cell update + ring buffers
__global__ __launch_bounds__(384) void k_gates(int t,
    const float* __restrict__ o, float* __restrict__ c_state,
    float* __restrict__ h_state, _Float16* __restrict__ h16,
    float* __restrict__ ring_h, float* __restrict__ ring_dis) {
  int b = blockIdx.x;
  int j = threadIdx.x;          // 0..383  (= l*C + c)
  __shared__ float s_fm[L_], s_im[L_];
  if (j == 0) {
    const float* ob = o + (size_t)b * NOP_;
    float m = fmaxf(ob[0], fmaxf(ob[1], ob[2]));
    float e0 = expf(ob[0] - m), e1 = expf(ob[1] - m), e2 = expf(ob[2] - m);
    float s = e0 + e1 + e2;
    s_fm[0] = e0 / s; s_fm[1] = (e0 + e1) / s; s_fm[2] = 1.0f;
    float m2 = fmaxf(ob[3], fmaxf(ob[4], ob[5]));
    float f0 = expf(ob[3] - m2), f1 = expf(ob[4] - m2), f2 = expf(ob[5] - m2);
    float s2 = f0 + f1 + f2;
    s_im[2] = f2 / s2; s_im[1] = (f1 + f2) / s2; s_im[0] = 1.0f;
    int w = t % KD_;
    ring_dis[(size_t)w * B_ + b] = 1.0f - (s_fm[0] + s_fm[1] + s_fm[2]) * (1.0f / 3.0f);
  }
  __syncthreads();
  int l = j / C_;
  const float* ob6 = o + (size_t)b * NOP_ + 2 * L_;
  float f  = sigmoidf_(ob6[j]);
  float ig = sigmoidf_(ob6[H_ + j]);
  float og = sigmoidf_(ob6[2 * H_ + j]);
  float ci = tanhf(ob6[3 * H_ + j]);
  float fme = s_fm[l], ime = s_im[l], ov = fme * ime;
  float cl = c_state[(size_t)b * H_ + j];
  float cn = ov * (f * cl + ig * ci) + (fme - ov) * cl + (ime - ov) * ci;
  float hn = og * tanhf(cn);
  c_state[(size_t)b * H_ + j] = cn;
  h_state[(size_t)b * H_ + j] = hn;
  h16[(size_t)b * H_ + j] = (_Float16)hn;
  int w = t % KD_;
  ring_h[((size_t)w * B_ + b) * H_ + j] = hn;
}

// ld = softmax(cumsum(dis)), lh16[b][h][k] = ring_h * ld, mean16 over k
__global__ __launch_bounds__(384) void k_ld_lh(int t,
    const float* __restrict__ ring_h, const float* __restrict__ ring_dis,
    _Float16* __restrict__ lh16, _Float16* __restrict__ mean16) {
  int b = blockIdx.x;
  __shared__ float s_ld[KD_];
  if (threadIdx.x == 0) {
    int w = t % KD_;
    float c[KD_], cum = 0.0f, mx = -1e30f;
    for (int k = 0; k < KD_; ++k) {
      int slot = (w + 1 + k) % KD_;
      cum += ring_dis[(size_t)slot * B_ + b];
      c[k] = cum;
      mx = fmaxf(mx, cum);
    }
    float s = 0.0f;
    for (int k = 0; k < KD_; ++k) { c[k] = expf(c[k] - mx); s += c[k]; }
    float inv = 1.0f / s;
    for (int k = 0; k < KD_; ++k) s_ld[k] = c[k] * inv;
  }
  __syncthreads();
  int h = threadIdx.x;
  int w = t % KD_;
  float mean = 0.0f;
  _Float16* dst = lh16 + ((size_t)b * H_ + h) * KD_;
#pragma unroll
  for (int k = 0; k < KD_; ++k) {
    int slot = (w + 1 + k) % KD_;
    float lv = ring_h[((size_t)slot * B_ + b) * H_ + h] * s_ld[k];
    dst[k] = (_Float16)lv;
    mean += lv;
  }
  mean16[(size_t)b * H_ + h] = (_Float16)(mean * (1.0f / KD_));
}

// t1 = relu(mean @ Wscale + bscale)   M=128 K=384 N=64 (4 tiles, one wave strip)
__global__ __launch_bounds__(32) void k_theme1(_Float16* __restrict__ t1,
    const _Float16* __restrict__ mean16, const _Float16* __restrict__ WscalePk,
    const float* __restrict__ bscale) {
  int m0 = blockIdx.y * 16;
  v8f acc[4] = {};
  for (int kt = 0; kt < (H_ >> 5); ++kt) {
    v16h a = load_a_tile(mean16 + (size_t)m0 * H_, H_, kt * 32);
#pragma unroll
    for (int i = 0; i < 4; ++i)
      acc[i] = WMMA_(a, load_b_packed(WscalePk, kt, NT_S, i), acc[i]);
  }
  int lane = threadIdx.x, g = lane >> 4, n = lane & 15;
#pragma unroll
  for (int i = 0; i < 4; ++i) {
    int col = i * 16 + n;
    float bias = bscale[col];
#pragma unroll
    for (int r = 0; r < 8; ++r)
      t1[(size_t)(m0 + r + 8 * g) * HS_ + col] = (_Float16)fmaxf(acc[i][r] + bias, 0.0f);
  }
}

// theme = sigmoid(t1 @ Wrescale + brescale)   M=128 K=64 N=384
__global__ __launch_bounds__(32) void k_theme2(float* __restrict__ theme,
    const _Float16* __restrict__ t1, const _Float16* __restrict__ WrescalePk,
    const float* __restrict__ brescale) {
  int nt0 = blockIdx.x * 4;      // gridDim.x = 6
  int m0 = blockIdx.y * 16;
  v8f acc[4] = {};
  for (int kt = 0; kt < (HS_ >> 5); ++kt) {
    v16h a = load_a_tile(t1 + (size_t)m0 * HS_, HS_, kt * 32);
#pragma unroll
    for (int i = 0; i < 4; ++i)
      acc[i] = WMMA_(a, load_b_packed(WrescalePk, kt, NT_H, nt0 + i), acc[i]);
  }
  int lane = threadIdx.x, g = lane >> 4, n = lane & 15;
#pragma unroll
  for (int i = 0; i < 4; ++i) {
    int col = (nt0 + i) * 16 + n;
    float bias = brescale[col];
#pragma unroll
    for (int r = 0; r < 8; ++r)
      theme[(size_t)(m0 + r + 8 * g) * H_ + col] = sigmoidf_(acc[i][r] + bias);
  }
}

// conv GEMM M=128 K=3840 N=384; out = hn + theme*(conv+bconv), written only when t==last[b]
__global__ __launch_bounds__(32) void k_conv(int t, float* __restrict__ sel,
    const _Float16* __restrict__ lh16, const _Float16* __restrict__ WconvPk,
    const float* __restrict__ bconv, const float* __restrict__ theme,
    const float* __restrict__ h_state, const int* __restrict__ last) {
  int nt0 = blockIdx.x * 4;      // gridDim.x = 6
  int m0 = blockIdx.y * 16;
  v8f acc[4] = {};
  for (int kt = 0; kt < ((H_ * KD_) >> 5); ++kt) {
    v16h a = load_a_tile(lh16 + (size_t)m0 * (H_ * KD_), H_ * KD_, kt * 32);
#pragma unroll
    for (int i = 0; i < 4; ++i)
      acc[i] = WMMA_(a, load_b_packed(WconvPk, kt, NT_H, nt0 + i), acc[i]);
  }
  int lane = threadIdx.x, g = lane >> 4, n = lane & 15;
#pragma unroll
  for (int i = 0; i < 4; ++i) {
    int col = (nt0 + i) * 16 + n;
    float bias = bconv[col];
#pragma unroll
    for (int r = 0; r < 8; ++r) {
      int row = m0 + r + 8 * g;
      if (last[row] == t) {
        float v = h_state[(size_t)row * H_ + col] +
                  theme[(size_t)row * H_ + col] * (acc[i][r] + bias);
        sel[(size_t)row * H_ + col] = v;
      }
    }
  }
}

// final FC (exact f32)
__global__ __launch_bounds__(128) void k_fc(float* __restrict__ out,
    const float* __restrict__ sel, const float* __restrict__ Wfc,
    const float* __restrict__ bfc) {
  int b = blockIdx.x, o = threadIdx.x;
  float acc = bfc[o];
  const float* s = sel + (size_t)b * H_;
  for (int h = 0; h < H_; ++h) acc += s[h] * Wfc[(size_t)h * OUT_ + o];
  out[(size_t)b * OUT_ + o] = acc;
}

// ---------------- host launch ----------------

extern "C" void kernel_launch(void* const* d_in, const int* in_sizes, int n_in,
                              void* d_out, int out_size, void* d_ws, size_t ws_size,
                              hipStream_t stream) {
  const int*   multihot = (const int*)  d_in[0];
  const float* emb      = (const float*)d_in[1];
  const float* Wk       = (const float*)d_in[2];
  const float* bk       = (const float*)d_in[3];
  const float* Wr       = (const float*)d_in[4];
  const float* br       = (const float*)d_in[5];
  const float* Wscale   = (const float*)d_in[6];
  const float* bscale   = (const float*)d_in[7];
  const float* Wrescale = (const float*)d_in[8];
  const float* brescale = (const float*)d_in[9];
  const float* Wconv    = (const float*)d_in[10];
  const float* bconv    = (const float*)d_in[11];
  const float* Wfc      = (const float*)d_in[12];
  const float* bfc      = (const float*)d_in[13];
  float* out = (float*)d_out;

  size_t off = 0;
  auto alloc = [&](size_t bytes) -> void* {
    void* p = (char*)d_ws + off;
    off += (bytes + 255) & ~(size_t)255;
    return p;
  };
  _Float16* x16       = (_Float16*)alloc((size_t)B_ * T_ * D_ * 2);
  int*      mask      = (int*)     alloc((size_t)B_ * T_ * 4);
  int*      last      = (int*)     alloc((size_t)B_ * 4);
  _Float16* WkPk      = (_Float16*)alloc((size_t)(D_ / 32) * NT_O * 512 * 2);
  _Float16* WrPk      = (_Float16*)alloc((size_t)(H_ / 32) * NT_O * 512 * 2);
  float*    bo        = (float*)   alloc((size_t)NOP_ * 4);
  _Float16* WscalePk  = (_Float16*)alloc((size_t)(H_ / 32) * NT_S * 512 * 2);
  _Float16* WrescalePk= (_Float16*)alloc((size_t)(HS_ / 32) * NT_H * 512 * 2);
  _Float16* WconvPk   = (_Float16*)alloc((size_t)(H_ * KD_ / 32) * NT_H * 512 * 2);
  float*    obuf      = (float*)   alloc((size_t)B_ * NOP_ * 4);
  float*    c_state   = (float*)   alloc((size_t)B_ * H_ * 4);
  float*    h_state   = (float*)   alloc((size_t)B_ * H_ * 4);
  _Float16* h16       = (_Float16*)alloc((size_t)B_ * H_ * 2);
  float*    ring_h    = (float*)   alloc((size_t)KD_ * B_ * H_ * 4);
  float*    ring_dis  = (float*)   alloc((size_t)KD_ * B_ * 4);
  _Float16* lh16      = (_Float16*)alloc((size_t)B_ * H_ * KD_ * 2);
  _Float16* mean16    = (_Float16*)alloc((size_t)B_ * H_ * 2);
  _Float16* t1_16     = (_Float16*)alloc((size_t)B_ * HS_ * 2);
  float*    theme     = (float*)   alloc((size_t)B_ * H_ * 4);
  float*    sel       = (float*)   alloc((size_t)B_ * H_ * 4);

  // --- init state ---
  k_zero_f32<<<96, 256, 0, stream>>>(c_state, B_ * H_);
  k_zero_f32<<<96, 256, 0, stream>>>(h_state, B_ * H_);
  k_zero_f16<<<96, 256, 0, stream>>>(h16, B_ * H_);
  k_zero_f32<<<480, 256, 0, stream>>>(ring_h, KD_ * B_ * H_);
  k_zero_f32<<<8, 256, 0, stream>>>(ring_dis, KD_ * B_);
  k_zero_f32<<<96, 256, 0, stream>>>(sel, B_ * H_);

  // --- weight prep: pack B operands into WMMA lane layout (f16, N padded) ---
  k_pack_b<<<512, 256, 0, stream>>>(WkPk, Wk, D_, NO_, NT_O);
  k_pack_b<<<1024, 256, 0, stream>>>(WrPk, Wr, H_, NO_, NT_O);
  k_pack_b<<<96, 256, 0, stream>>>(WscalePk, Wscale, H_, HS_, NT_S);
  k_pack_b<<<96, 256, 0, stream>>>(WrescalePk, Wrescale, HS_, H_, NT_H);
  k_pack_bT<<<1024, 256, 0, stream>>>(WconvPk, Wconv, H_ * KD_, H_, NT_H);
  k_build_bo<<<(NOP_ + 255) / 256, 256, 0, stream>>>(bo, bk, br,
      Wk + (size_t)D_ * NO_, Wr + (size_t)H_ * NO_);

  // --- embedding + mask + last ---
  k_embed<<<B_ * T_, 128, 0, stream>>>(x16, mask, multihot, emb);
  k_last<<<1, 128, 0, stream>>>(last, mask);

  // --- sequential scan ---
  for (int t = 0; t < T_; ++t) {
    k_gemm_o<<<dim3(25, 8), 32, 0, stream>>>(obuf, x16 + (size_t)t * D_, T_ * D_,
                                             WkPk, h16, WrPk, bo);
    k_gates<<<B_, 384, 0, stream>>>(t, obuf, c_state, h_state, h16, ring_h, ring_dis);
    k_ld_lh<<<B_, 384, 0, stream>>>(t, ring_h, ring_dis, lh16, mean16);
    k_theme1<<<dim3(1, 8), 32, 0, stream>>>(t1_16, mean16, WscalePk, bscale);
    k_theme2<<<dim3(6, 8), 32, 0, stream>>>(theme, t1_16, WrescalePk, brescale);
    k_conv<<<dim3(6, 8), 32, 0, stream>>>(t, sel, lh16, WconvPk, bconv, theme,
                                          h_state, last);
  }

  // --- final FC ---
  k_fc<<<B_, 128, 0, stream>>>(out, sel, Wfc, bfc);
}
